// BallMSA_360777253452
// MI455X (gfx1250) — compile-verified, matrix operand order checked by reading, not run on hardware
//
#include <hip/hip_runtime.h>
#include <hip/hip_bf16.h>

typedef __bf16 bf16;
typedef __attribute__((ext_vector_type(16))) __bf16 v16bf;
typedef __attribute__((ext_vector_type(8)))  __bf16 bf16x8;
typedef __attribute__((ext_vector_type(8)))  float  v8f;
typedef __attribute__((ext_vector_type(4)))  float  f32x4;
typedef int v4i_vs __attribute__((vector_size(16)));   // matches builtin param type

static constexpr int kDIM = 256;
static constexpr int kM   = 64;
static constexpr float kSCALE = 0.17677669529663687f; // 1/sqrt(32)

// LDS byte offsets (all 16B aligned; total 274432 B < 320KB/WGP)
static constexpr unsigned OFF_REL  = 0;        // float [64][4]
static constexpr unsigned OFF_DIST = 1024;     // float [64][68]
static constexpr unsigned OFF_X    = 18432;    // bf16  [64][264]
static constexpr unsigned OFF_Q    = 52224;    // bf16  [8][64][32]
static constexpr unsigned OFF_K    = 84992;    // bf16  [8][64][32]
static constexpr unsigned OFF_V    = 117760;   // bf16  [8][32][80]  (V^T per head)
static constexpr unsigned OFF_P    = 158720;   // bf16  [8][64][80]; aliased early as raw-x f32 staging (64KB)
static constexpr unsigned OFF_O    = 240640;   // bf16  [64][264]
static constexpr unsigned LDS_BYTES = 274432;

#if __has_builtin(__builtin_amdgcn_global_load_async_to_lds_b128)
#define HAS_ASYNC_LDS 1
#else
#define HAS_ASYNC_LDS 0
#endif

__device__ __forceinline__ v8f zero8() {
  v8f z;
#pragma unroll
  for (int i = 0; i < 8; ++i) z[i] = 0.0f;
  return z;
}

// Build a 16-bit WMMA A/B fragment from a row-major bf16 LDS matrix.
// lane<16 : row = row0+lane,    K chunks {k0..k0+7}, {k0+16..k0+23}
// lane>=16: row = row0+lane-16, K chunks {k0+8..k0+15}, {k0+24..k0+31}
__device__ __forceinline__ v16bf frag_ld_lds(const bf16* base, int row0, int stride,
                                             int k0, int lane) {
  int r = row0 + (lane & 15);
  int half8 = ((lane >> 4) & 1) * 8;
  const bf16* p = base + r * stride + k0 + half8;
  bf16x8 lo = *(const bf16x8*)(p);
  bf16x8 hi = *(const bf16x8*)(p + 16);
  return __builtin_shufflevector(lo, hi, 0,1,2,3,4,5,6,7,8,9,10,11,12,13,14,15);
}

// Same fragment, but from a per-lane fp32 row in global memory (weights: L2-resident),
// converted to bf16 in-register.  rowp already points at (row, k0).
__device__ __forceinline__ v16bf frag_ld_gf32(const float* rowp, int lane) {
  int half8 = ((lane >> 4) & 1) * 8;
  const f32x4* p = (const f32x4*)(rowp + half8);
  f32x4 a0 = p[0];
  f32x4 a1 = p[1];
  const f32x4* q = (const f32x4*)(rowp + half8 + 16);
  f32x4 b0 = q[0];
  f32x4 b1 = q[1];
  v16bf r;
#pragma unroll
  for (int i = 0; i < 4; ++i) {
    r[i]      = (bf16)a0[i];
    r[4 + i]  = (bf16)a1[i];
    r[8 + i]  = (bf16)b0[i];
    r[12 + i] = (bf16)b1[i];
  }
  return r;
}

__device__ __forceinline__ v8f wmma_bf16(v16bf a, v16bf b, v8f c) {
  return __builtin_amdgcn_wmma_f32_16x16x32_bf16(false, a, false, b, (short)0, c,
                                                 false, false);
}

__global__ __launch_bounds__(256) void ballmsa_fused_kernel(
    const float* __restrict__ x, const float* __restrict__ pos,
    const float* __restrict__ w_qkv, const float* __restrict__ b_qkv,
    const float* __restrict__ w_pe, const float* __restrict__ b_pe,
    const float* __restrict__ w_proj, const float* __restrict__ b_proj,
    const float* __restrict__ sigma_att, float* __restrict__ out) {
  extern __shared__ __align__(16) char smem[];
  float* s_rel  = (float*)(smem + OFF_REL);
  float* s_dist = (float*)(smem + OFF_DIST);
  bf16*  s_x    = (bf16*)(smem + OFF_X);
  bf16*  s_q    = (bf16*)(smem + OFF_Q);
  bf16*  s_k    = (bf16*)(smem + OFF_K);
  bf16*  s_v    = (bf16*)(smem + OFF_V);
  bf16*  s_p    = (bf16*)(smem + OFF_P);
  bf16*  s_o    = (bf16*)(smem + OFF_O);
  float* s_xraw = (float*)(smem + OFF_P);   // alias: raw x staging, dead before P is live

  const int tid  = threadIdx.x;
  const int lane = tid & 31;
  const int wave = tid >> 5;        // wave w <-> head w
  const int hi8  = ((lane >> 4) & 1) * 8;
  const int nloc = lane & 15;
  const size_t tok0 = (size_t)blockIdx.x * kM;

#if HAS_ASYNC_LDS
  // Async-copy the ball's x tile (64x256 f32 = 64KB) cache->LDS while we do the
  // position preamble.  16 b128 transfers per thread, tracked by ASYNCcnt.
  {
    const float* gx = x + tok0 * kDIM;
#pragma unroll
    for (int it = 0; it < 16; ++it) {
      int idx4 = tid + it * 256;    // float4 index within the 4096-float4 tile
      __builtin_amdgcn_global_load_async_to_lds_b128(
          (v4i_vs*)(gx + idx4 * 4),
          (v4i_vs*)(s_xraw + idx4 * 4),
          0, 0);
    }
  }
#endif

  // Warm L2 with this wave's w_qkv slice (emits global_prefetch_b8).
  for (int r = lane; r < 96; r += 32)
    __builtin_prefetch(w_qkv + ((size_t)wave * 96 + r) * kDIM, 0, 2);

  // ---------- phase 0: rel positions, pairwise distances, x' ----------
  if (tid < kM) {
    s_rel[tid * 4 + 0] = pos[(tok0 + tid) * 3 + 0];
    s_rel[tid * 4 + 1] = pos[(tok0 + tid) * 3 + 1];
    s_rel[tid * 4 + 2] = pos[(tok0 + tid) * 3 + 2];
  }
  __syncthreads();
  float mx = 0.f, my = 0.f, mz = 0.f;
  for (int i = 0; i < kM; ++i) {
    mx += s_rel[i * 4 + 0]; my += s_rel[i * 4 + 1]; mz += s_rel[i * 4 + 2];
  }
  mx *= (1.0f / kM); my *= (1.0f / kM); mz *= (1.0f / kM);
  __syncthreads();
  if (tid < kM) {
    s_rel[tid * 4 + 0] -= mx; s_rel[tid * 4 + 1] -= my; s_rel[tid * 4 + 2] -= mz;
  }

#if HAS_ASYNC_LDS
#if __has_builtin(__builtin_amdgcn_s_wait_asynccnt)
  __builtin_amdgcn_s_wait_asynccnt(0);
#else
  asm volatile("s_wait_asynccnt 0" ::: "memory");
#endif
#endif
  __syncthreads();   // rel final + (async x staging visible to all waves)

  for (int i = tid; i < kM * kM; i += 256) {
    int m = i >> 6, k2 = i & 63;
    float dx = s_rel[m * 4 + 0] - s_rel[k2 * 4 + 0];
    float dy = s_rel[m * 4 + 1] - s_rel[k2 * 4 + 1];
    float dz = s_rel[m * 4 + 2] - s_rel[k2 * 4 + 2];
    s_dist[m * 68 + k2] = sqrtf(fmaxf(dx * dx + dy * dy + dz * dz, 0.0f));
  }
  for (int i = tid; i < kM * kDIM; i += 256) {
    int m = i >> 8, c = i & 255;
#if HAS_ASYNC_LDS
    float xv = s_xraw[m * kDIM + c];
#else
    float xv = x[(tok0 + m) * kDIM + c];
#endif
    float v = xv
            + s_rel[m * 4 + 0] * w_pe[c * 3 + 0]
            + s_rel[m * 4 + 1] * w_pe[c * 3 + 1]
            + s_rel[m * 4 + 2] * w_pe[c * 3 + 2]
            + b_pe[c];
    s_x[m * 264 + c] = (bf16)v;
  }
  __syncthreads();

  // ---------- phase 1: QKV = x' @ w_qkv^T  (wave h -> channels [96h, 96h+96)) ----
  const int h = wave;
  for (int ct = 0; ct < 6; ++ct) {
    const int c   = h * 96 + ct * 16 + nloc;        // output channel for lane column
    const float bq = b_qkv[c];
    const int e   = (ct * 16 + nloc) / 3;           // (c % 96) / 3
    const int sel = (ct * 16 + nloc) % 3;           // c % 3  (0=q,1=k,2=v)
    v8f acc[4];
#pragma unroll
    for (int rt = 0; rt < 4; ++rt) acc[rt] = zero8();
    for (int kk = 0; kk < 8; ++kk) {
      v16bf bfrag = frag_ld_gf32(w_qkv + (size_t)c * kDIM + kk * 32, lane);
#pragma unroll
      for (int rt = 0; rt < 4; ++rt) {
        v16bf afrag = frag_ld_lds(s_x, rt * 16, 264, kk * 32, lane);
        acc[rt] = wmma_bf16(afrag, bfrag, acc[rt]);
      }
    }
    bf16* dstq = s_q + h * (64 * 32);
    bf16* dstk = s_k + h * (64 * 32);
    bf16* dstv = s_v + h * (32 * 80);               // transposed: Vt[e][row]
#pragma unroll
    for (int rt = 0; rt < 4; ++rt) {
#pragma unroll
      for (int vi = 0; vi < 8; ++vi) {
        int row = rt * 16 + vi + hi8;
        bf16 bv = (bf16)(acc[rt][vi] + bq);
        if (sel == 0)      dstq[row * 32 + e] = bv;
        else if (sel == 1) dstk[row * 32 + e] = bv;
        else               dstv[e * 80 + row] = bv;
      }
    }
  }
  __syncthreads();

  // ---------- phase 2: attention, wave h owns head h ----------
  {
    const bf16* Q  = s_q + h * (64 * 32);
    const bf16* K  = s_k + h * (64 * 32);
    const bf16* Vt = s_v + h * (32 * 80);
    bf16* P = s_p + h * (64 * 80);
    const float sigma = sigma_att[h];
    for (int mt = 0; mt < 4; ++mt) {
      v16bf qf = frag_ld_lds(Q, mt * 16, 32, 0, lane);
      v8f sc[4];
#pragma unroll
      for (int nt = 0; nt < 4; ++nt) {
        v16bf kf = frag_ld_lds(K, nt * 16, 32, 0, lane); // B[e][n] = K[n][e]
        sc[nt] = wmma_bf16(qf, kf, zero8());
      }
      // scale + distance bias
#pragma unroll
      for (int nt = 0; nt < 4; ++nt) {
        int col = nt * 16 + nloc;
#pragma unroll
        for (int vi = 0; vi < 8; ++vi) {
          int row = mt * 16 + vi + hi8;
          sc[nt][vi] = sc[nt][vi] * kSCALE + sigma * s_dist[row * 68 + col];
        }
      }
      // row softmax: 16-lane reductions (row groups split at lane 16; xor<=8 stays inside)
#pragma unroll
      for (int vi = 0; vi < 8; ++vi) {
        float rmx = fmaxf(fmaxf(sc[0][vi], sc[1][vi]), fmaxf(sc[2][vi], sc[3][vi]));
#pragma unroll
        for (int o = 1; o < 16; o <<= 1) rmx = fmaxf(rmx, __shfl_xor(rmx, o, 32));
        float rs = 0.f;
#pragma unroll
        for (int nt = 0; nt < 4; ++nt) {
          float p = __expf(sc[nt][vi] - rmx);
          sc[nt][vi] = p;
          rs += p;
        }
#pragma unroll
        for (int o = 1; o < 16; o <<= 1) rs += __shfl_xor(rs, o, 32);
        float inv = 1.0f / rs;
#pragma unroll
        for (int nt = 0; nt < 4; ++nt) sc[nt][vi] *= inv;
      }
#pragma unroll
      for (int nt = 0; nt < 4; ++nt) {
        int col = nt * 16 + nloc;
#pragma unroll
        for (int vi = 0; vi < 8; ++vi) {
          int row = mt * 16 + vi + hi8;
          P[row * 80 + col] = (bf16)sc[nt][vi];
        }
      }
    }
    // O = P @ V  (LDS producer/consumer within one wave: DS ops are in-order)
    for (int mt = 0; mt < 4; ++mt) {
      v8f oacc[2];
      oacc[0] = zero8(); oacc[1] = zero8();
#pragma unroll
      for (int kk = 0; kk < 2; ++kk) {
        v16bf pf = frag_ld_lds(P, mt * 16, 80, kk * 32, lane);
#pragma unroll
        for (int et = 0; et < 2; ++et) {
          v16bf vf = frag_ld_lds(Vt, et * 16, 80, kk * 32, lane); // B[k][e] = Vt[e][k]
          oacc[et] = wmma_bf16(pf, vf, oacc[et]);
        }
      }
#pragma unroll
      for (int et = 0; et < 2; ++et) {
#pragma unroll
        for (int vi = 0; vi < 8; ++vi) {
          int row = mt * 16 + vi + hi8;
          s_o[row * 264 + h * 32 + et * 16 + nloc] = (bf16)oacc[et][vi];
        }
      }
    }
  }
  __syncthreads();

  // ---------- phase 3: out = O @ w_proj^T + b_proj ----------
  for (int cti = 0; cti < 2; ++cti) {
    const int c = (wave * 2 + cti) * 16 + nloc;
    const float bp = b_proj[c];
    v8f acc[4];
#pragma unroll
    for (int rt = 0; rt < 4; ++rt) acc[rt] = zero8();
    for (int kk = 0; kk < 8; ++kk) {
      v16bf bfrag = frag_ld_gf32(w_proj + (size_t)c * kDIM + kk * 32, lane);
#pragma unroll
      for (int rt = 0; rt < 4; ++rt) {
        v16bf afrag = frag_ld_lds(s_o, rt * 16, 264, kk * 32, lane);
        acc[rt] = wmma_bf16(afrag, bfrag, acc[rt]);
      }
    }
#pragma unroll
    for (int rt = 0; rt < 4; ++rt) {
#pragma unroll
      for (int vi = 0; vi < 8; ++vi) {
        int row = rt * 16 + vi + hi8;
        out[(tok0 + row) * kDIM + c] = acc[rt][vi] + bp;
      }
    }
  }
}

extern "C" void kernel_launch(void* const* d_in, const int* in_sizes, int n_in,
                              void* d_out, int out_size, void* d_ws, size_t ws_size,
                              hipStream_t stream) {
  (void)n_in; (void)out_size; (void)d_ws; (void)ws_size;
  const float* x      = (const float*)d_in[0];
  const float* pos    = (const float*)d_in[1];
  const float* w_qkv  = (const float*)d_in[2];
  const float* b_qkv  = (const float*)d_in[3];
  const float* w_pe   = (const float*)d_in[4];
  const float* b_pe   = (const float*)d_in[5];
  const float* w_proj = (const float*)d_in[6];
  const float* b_proj = (const float*)d_in[7];
  const float* sigma  = (const float*)d_in[8];
  float* out = (float*)d_out;

  const int n_balls = in_sizes[1] / (64 * 3);  // pos holds N*3 floats, M=64
  hipLaunchKernelGGL(ballmsa_fused_kernel, dim3(n_balls), dim3(256), LDS_BYTES,
                     stream, x, pos, w_qkv, b_qkv, w_pe, b_pe, w_proj, b_proj,
                     sigma, out);
}
